// HungarianMatcher_4466765988424
// MI455X (gfx1250) — matrix operand order repaired and checked by reading, not compiled, not run on hardware
//
#include <hip/hip_runtime.h>

#define B 4
#define H 128
#define W 128
#define Q 512
#define E 128
#define P (H*W)            // 16384
#define SPLITK 8
#define PSEG (P/SPLITK)    // 2048
#define KC 32
#define STRA 40            // padded LDS row stride (elements), 80B => 16B aligned rows

typedef __attribute__((ext_vector_type(16))) __bf16 v16bf;
typedef __attribute__((ext_vector_type(8)))  float  v8f;

// ---- workspace layout (in floats) ----
#define ROWMAX_OFF 0
#define ROWRCP_OFF (B*P)                 // 65536
#define PCS_OFF    (2*B*P)               // 131072  portions col-sum (B,Q)
#define SCS_OFF    (PCS_OFF + B*Q)       // 133120  sg col-sum (B,E)
#define NNZ_OFF    (SCS_OFF + B*E)       // 133632  (B)
#define ACCM_OFF   135168                // mask GEMM accum (B,Q,E)
#define ACCD_OFF   (ACCM_OFF + B*Q*E)    // dice GEMM accum (B,Q,E)
#define WS_FLOATS  (ACCD_OFF + B*Q*E)

__device__ __forceinline__ float wred_max(float v) {
    for (int o = 16; o; o >>= 1) v = fmaxf(v, __shfl_xor(v, o, 32));
    return v;
}
__device__ __forceinline__ float wred_sum(float v) {
    for (int o = 16; o; o >>= 1) v += __shfl_xor(v, o, 32);
    return v;
}
// returns softplus(x); sets sn = softplus(-x).
// e = exp(-|x|) in (0,1] => 1+e in (1,2]: hardware v_log_f32 is plenty accurate
// (results are rounded to bf16 downstream anyway).
__device__ __forceinline__ float softplus_pair(float x, float& sn) {
    float l = __logf(1.0f + __expf(-fabsf(x)));
    sn = l + fmaxf(-x, 0.f);
    return l + fmaxf(x, 0.f);
}

// ---------------- kernel 1: per-pixel masked softmax stats over Q ----------------
__global__ void k_rowstats(const float* __restrict__ ml, const unsigned char* __restrict__ mp,
                           float* __restrict__ rmax, float* __restrict__ rrcp) {
    int wv = threadIdx.x >> 5, lane = threadIdx.x & 31;
    long rp = (long)blockIdx.x * 8 + wv;           // pixel id in [0, B*P)
    const float* row = ml + rp * (long)Q;
    const unsigned char* mrow = mp + rp * (long)Q;
    float x[16]; unsigned mbits = 0; float mx = -1e30f;
    #pragma unroll
    for (int i = 0; i < 16; ++i) {
        int q = lane + (i << 5);
        x[i] = row[q];
        if (mrow[q]) { mbits |= 1u << i; mx = fmaxf(mx, x[i]); }
    }
    mx = wred_max(mx);
    float s = 0.f;
    #pragma unroll
    for (int i = 0; i < 16; ++i)
        if ((mbits >> i) & 1u) s += __expf(x[i] - mx);
    s = wred_sum(s);
    if (lane == 0) { rmax[rp] = mx; rrcp[rp] = (s > 0.f) ? 1.f / s : 0.f; }
}

// ---------------- kernel 2: column sums (portions over p, sg over p, nnz) ----------------
__global__ void k_colsums(const float* __restrict__ ml, const unsigned char* __restrict__ mp,
                          const float* __restrict__ sv, const unsigned char* __restrict__ smk,
                          const float* __restrict__ rmax, const float* __restrict__ rrcp,
                          float* __restrict__ pcs, float* __restrict__ scs, float* __restrict__ nnz) {
    int b = blockIdx.x >> 7;
    int p0 = (blockIdx.x & 127) << 7;              // 128 pixels per block
    int t = threadIdx.x;
    float aq0 = 0.f, aq1 = 0.f, asg = 0.f, atg = 0.f;
    long base = (long)b * P + p0;
    for (int pp = 0; pp < 128; ++pp) {
        long rp = base + pp;
        float rm = rmax[rp], rr = rrcp[rp];
        long mb = rp * (long)Q;
        { int q = t;        float x = ml[mb + q]; if (mp[mb + q]) aq0 += __expf(x - rm) * rr; }
        { int q = t + 256;  float x = ml[mb + q]; if (mp[mb + q]) aq1 += __expf(x - rm) * rr; }
        if (t < E) {
            long sb = rp * (long)E + t;
            float s = smk[sb] ? sv[sb] : 0.f;
            asg += s;
            atg += (s != 0.f) ? 1.f : 0.f;
        }
    }
    unsafeAtomicAdd(&pcs[b * Q + t],       aq0);
    unsafeAtomicAdd(&pcs[b * Q + t + 256], aq1);
    if (t < E) {
        unsafeAtomicAdd(&scs[b * E + t], asg);
        float wsum = wred_sum(atg);                // waves 0..3 are fully active here
        if ((t & 31) == 0) unsafeAtomicAdd(&nnz[b], wsum);
    }
}

// ---------------- kernel 3: fused bf16 WMMA GEMMs with split-K ----------------
// tile: 64 q x 64 e, 8 waves; wave w: q-subtile (w&3), e-subtiles {2*(w>>2), 2*(w>>2)+1}
__global__ void __launch_bounds__(256)
k_gemm(const float* __restrict__ ml, const unsigned char* __restrict__ mp,
       const float* __restrict__ sv, const unsigned char* __restrict__ smk,
       const float* __restrict__ rmax, const float* __restrict__ rrcp,
       float* __restrict__ accM, float* __restrict__ accD) {
    __shared__ __bf16 sApos[64 * STRA];
    __shared__ __bf16 sAneg[64 * STRA];
    __shared__ __bf16 sApor[64 * STRA];
    __shared__ __bf16 sBt  [64 * STRA];
    __shared__ __bf16 sB1  [64 * STRA];
    __shared__ __bf16 sBs  [64 * STRA];

    int bi = blockIdx.x;
    int qb = bi & 7;            // Q/64 = 8
    int eb = (bi >> 3) & 1;     // E/64 = 2
    int b  = (bi >> 4) & 3;
    int ks = bi >> 6;           // split-K segment
    int qbase = qb << 6, ebase = eb << 6;
    int pstart = ks * PSEG, pend = pstart + PSEG;

    int tid = threadIdx.x;
    int qf = tid & 63;          // fill column (q or e within tile)
    int kq = tid >> 6;          // 0..3

    int lane = tid & 31, wv = tid >> 5;
    int qi = wv & 3, esel = wv >> 2;
    int mrow = lane & 15;
    int aoff = (qi * 16 + mrow) * STRA + ((lane < 16) ? 0 : 8);

    v8f accMf[2] = { {}, {} };
    v8f accDf[2] = { {}, {} };

    union Frag { v16bf v; uint4 u[2]; };

    for (int p0 = pstart; p0 < pend; p0 += KC) {
        // ---- cooperative fill: compute pos/neg/portions + targ/(1-targ)/sg in bf16 ----
        #pragma unroll
        for (int i = 0; i < 8; ++i) {
            int kp = kq + (i << 2);
            long rp = (long)b * P + p0 + kp;
            float rm = rmax[rp], rr = rrcp[rp];
            long ab = rp * (long)Q + qbase + qf;
            float x = ml[ab];
            bool m = mp[ab] != 0;
            float sn; float sp = softplus_pair(x, sn);
            float por = m ? __expf(x - rm) * rr : 0.f;
            int so = qf * STRA + kp;
            sApos[so] = (__bf16)(m ? sn : 0.f);
            sAneg[so] = (__bf16)(m ? sp : 0.f);
            sApor[so] = (__bf16)por;
            long bb = rp * (long)E + ebase + qf;
            float s = (smk[bb] != 0) ? sv[bb] : 0.f;
            float tg = (s != 0.f) ? 1.f : 0.f;
            sBt[so] = (__bf16)tg;
            sB1[so] = (__bf16)(1.f - tg);
            sBs[so] = (__bf16)s;
        }
        if (p0 + KC < pend) { // pull next chunk toward L2/WGP$ (global_prefetch_b8)
            __builtin_prefetch(&ml[((long)b * P + p0 + KC + kq) * (long)Q + qbase + qf], 0, 1);
            __builtin_prefetch(&sv[((long)b * P + p0 + KC + kq) * (long)E + ebase + qf], 0, 1);
        }
        __syncthreads();

        // ---- fragment loads (ISA 16-bit A/B layouts) + WMMA ----
        Frag fp, fn, fo;
        fp.u[0] = *(const uint4*)&sApos[aoff]; fp.u[1] = *(const uint4*)&sApos[aoff + 16];
        fn.u[0] = *(const uint4*)&sAneg[aoff]; fn.u[1] = *(const uint4*)&sAneg[aoff + 16];
        fo.u[0] = *(const uint4*)&sApor[aoff]; fo.u[1] = *(const uint4*)&sApor[aoff + 16];
        #pragma unroll
        for (int j = 0; j < 2; ++j) {
            int es = esel * 2 + j;
            int boff = (es * 16 + mrow) * STRA + ((lane < 16) ? 0 : 8);
            Frag bt, b1, bs;
            bt.u[0] = *(const uint4*)&sBt[boff]; bt.u[1] = *(const uint4*)&sBt[boff + 16];
            b1.u[0] = *(const uint4*)&sB1[boff]; b1.u[1] = *(const uint4*)&sB1[boff + 16];
            bs.u[0] = *(const uint4*)&sBs[boff]; bs.u[1] = *(const uint4*)&sBs[boff + 16];
            accMf[j] = __builtin_amdgcn_wmma_f32_16x16x32_bf16(false, fp.v, false, bt.v, (short)0, accMf[j], false, false);
            accMf[j] = __builtin_amdgcn_wmma_f32_16x16x32_bf16(false, fn.v, false, b1.v, (short)0, accMf[j], false, false);
            accDf[j] = __builtin_amdgcn_wmma_f32_16x16x32_bf16(false, fo.v, false, bs.v, (short)0, accDf[j], false, false);
        }
        __syncthreads();
    }

    // ---- split-K reduction via f32 atomics (C/D layout: VGPR r -> M=r+8*(lane>>4), N=lane&15) ----
    int col = lane & 15, rb = (lane >> 4) * 8;
    #pragma unroll
    for (int j = 0; j < 2; ++j) {
        int es = esel * 2 + j;
        #pragma unroll
        for (int r = 0; r < 8; ++r) {
            int q = qbase + qi * 16 + rb + r;
            int e = ebase + es * 16 + col;
            long o = ((long)(b * Q + q)) * E + e;
            unsafeAtomicAdd(&accM[o], accMf[j][r]);
            unsafeAtomicAdd(&accD[o], accDf[j][r]);
        }
    }
}

// ---------------- kernel 4: epilogue (class + mask + dice + huber + nll + like) ----------------
__global__ void k_epilogue(const float* __restrict__ accM, const float* __restrict__ accD,
                           const float* __restrict__ pcs, const float* __restrict__ scs,
                           const float* __restrict__ nnz,
                           const float* __restrict__ pcl, const float* __restrict__ ppos,
                           const float* __restrict__ chol, const float* __restrict__ tpos,
                           const float* __restrict__ isz, float* __restrict__ out) {
    int idx = blockIdx.x * blockDim.x + threadIdx.x;   // (b*Q + q)*E + e
    int e = idx & (E - 1);
    int q = (idx >> 7) & (Q - 1);
    int b = idx >> 16;
    float mask_cost = accM[idx] / nnz[b];
    float dice = 1.f - (2.f * accD[idx] + 1.f) / (pcs[b * Q + q] + scs[b * E + e] + 1.f);
    float cl = pcl[b * Q + q];
    float cls = __logf(1.0f + __expf(-fabsf(cl))) + fmaxf(-cl, 0.f);   // softplus(-cl)
    float px = ppos[(b * Q + q) * 2], py = ppos[(b * Q + q) * 2 + 1];
    float tx = tpos[(b * E + e) * 2], ty = tpos[(b * E + e) * 2 + 1];
    float d0 = px - tx, d1 = py - ty;
    float a0 = fabsf(d0), a1 = fabsf(d1);
    float h0 = (a0 < 1.f) ? 0.5f * d0 * d0 : a0 - 0.5f;
    float h1 = (a1 < 1.f) ? 0.5f * d1 * d1 : a1 - 0.5f;
    float dist = 0.5f * (h0 + h1);
    float sx = isz[b * 2], sy = isz[b * 2 + 1];
    const float* L = chol + (long)(b * Q + q) * 4;
    float L00 = L[0], L10 = L[2], L11 = L[3];
    float dx = (tx - px) * sx, dy = (ty - py) * sy;
    float z0 = dx / L00;
    float z1 = (dy - L10 * z0) / L11;
    float logp = -0.5f * (z0 * z0 + z1 * z1) - __logf(L00 * L11) - 1.8378770664093453f;
    out[idx] = cls + mask_cost + dice + dist + (-logp) + (1.f - __expf(logp));
}

extern "C" void kernel_launch(void* const* d_in, const int* in_sizes, int n_in,
                              void* d_out, int out_size, void* d_ws, size_t ws_size,
                              hipStream_t stream) {
    const float*         ml   = (const float*)d_in[0];
    const unsigned char* mp   = (const unsigned char*)d_in[1];
    const float*         sv   = (const float*)d_in[2];
    const unsigned char* smk  = (const unsigned char*)d_in[3];
    const float*         pcl  = (const float*)d_in[4];
    const float*         ppos = (const float*)d_in[5];
    const float*         chol = (const float*)d_in[6];
    const float*         tpos = (const float*)d_in[7];
    const float*         isz  = (const float*)d_in[8];
    float* out = (float*)d_out;
    float* ws  = (float*)d_ws;

    hipMemsetAsync(d_ws, 0, (size_t)WS_FLOATS * sizeof(float), stream);

    k_rowstats<<<(B * P) / 8, 256, 0, stream>>>(ml, mp, ws + ROWMAX_OFF, ws + ROWRCP_OFF);

    k_colsums<<<B * (P / 128), 256, 0, stream>>>(ml, mp, sv, smk,
                                                 ws + ROWMAX_OFF, ws + ROWRCP_OFF,
                                                 ws + PCS_OFF, ws + SCS_OFF, ws + NNZ_OFF);

    k_gemm<<<(Q / 64) * (E / 64) * B * SPLITK, 256, 0, stream>>>(ml, mp, sv, smk,
                                                                 ws + ROWMAX_OFF, ws + ROWRCP_OFF,
                                                                 ws + ACCM_OFF, ws + ACCD_OFF);

    k_epilogue<<<(B * Q * E) / 256, 256, 0, stream>>>(ws + ACCM_OFF, ws + ACCD_OFF,
                                                      ws + PCS_OFF, ws + SCS_OFF, ws + NNZ_OFF,
                                                      pcl, ppos, chol, tpos, isz, out);
}
